// CodebookLinear_6751688589354
// MI455X (gfx1250) — compile-verified, matrix operand order checked by reading, not run on hardware
//
#include <hip/hip_runtime.h>
#include <hip/hip_bf16.h>

// CodebookLinear as a bf16-WMMA GEMM with on-the-fly dequantization.
// M=8192, N=4096, K=4096.  Compute-bound (AI ~1000 FLOP/B) -> bf16 WMMA,
// f32 accumulate.  64x64 per-wave tile: 16 WMMAs per 8 fragment loads.

typedef __attribute__((ext_vector_type(16))) __bf16 v16bf;
typedef __attribute__((ext_vector_type(8)))  __bf16 v8bf;
typedef __attribute__((ext_vector_type(4)))  __bf16 v4bf;
typedef __attribute__((ext_vector_type(8)))  float  v8f;

#define OUT_F   4096
#define IN_F    4096
#define KCODES  256
#define BD      16          // codebook block dim
#define MROWS   8192        // 4 * 2048
#define MT      128         // block M tile
#define NT      256         // block N tile
#define KT      32          // K per WMMA / per pipeline step
#define PITCH   40          // bf16 elems per LDS row (80 B, breaks 64-bank aliasing)
#define NKSTEP  (IN_F / KT) // 128

__global__ __launch_bounds__(256)
void codebook_linear_wmma(const float* __restrict__ X,
                          const float* __restrict__ C,     // [256,16] codebook
                          const int*   __restrict__ L,     // [OUT_F * IN_F/16] labels
                          const float* __restrict__ B,     // [OUT_F] bias
                          float* __restrict__ O)           // [MROWS, OUT_F]
{
    __shared__ alignas(16) __bf16 sCent[KCODES * BD];      //  8 KB
    __shared__ alignas(16) __bf16 sX[2][MT * PITCH];       // 2 x 10 KB
    __shared__ alignas(16) __bf16 sW[2][NT * PITCH];       // 2 x 20 KB

    const int tid   = threadIdx.x;
    const int lane  = tid & 31;
    const int wave  = tid >> 5;
    const int wm    = wave & 1;    // 2 M-slabs of 64 rows
    const int wn    = wave >> 1;   // 4 N-slabs of 64 cols
    const int nbase = blockIdx.x * NT;
    const int mbase = blockIdx.y * MT;

    // ---- one-time: codebook f32 -> bf16 into LDS (256 threads, 1 row each) ----
    {
        const float4* src = (const float4*)(C + tid * BD);
#pragma unroll
        for (int i = 0; i < 4; ++i) {
            float4 v = src[i];
            v4bf w;
            w[0] = (__bf16)v.x; w[1] = (__bf16)v.y;
            w[2] = (__bf16)v.z; w[3] = (__bf16)v.w;
            *(v4bf*)&sCent[tid * BD + i * 4] = w;
        }
    }

    // x-tile copy map: 128 rows x 32 k f32 = 16 f32/thread (4x float4)
    int xrow[4], xcol[4];
#pragma unroll
    for (int i = 0; i < 4; ++i) {
        int idx = tid + i * 256;            // 0..1023
        xrow[i] = idx >> 3;                 // 0..127
        xcol[i] = (idx & 7) * 4;            // float4 column
    }
    // w-tile gather map: 256 rows x 2 label-blocks = 512 labels, 2/thread
    const int wrow = tid >> 1;              // 0..127 (+128 on second rep)
    const int wblk = tid & 1;               // 16-wide label block within K step

    auto fetchX = [&](int ks, float4* gx) {
        const float* base = X + (size_t)mbase * IN_F + ks * KT;
#pragma unroll
        for (int i = 0; i < 4; ++i)
            gx[i] = *(const float4*)(base + (size_t)xrow[i] * IN_F + xcol[i]);
    };
    auto fetchW = [&](int ks, v8bf* lo, v8bf* hi) {
#pragma unroll
        for (int rep = 0; rep < 2; ++rep) {
            int n   = rep * 128 + wrow;
            int lbl = L[(size_t)(nbase + n) * (IN_F / BD) + ks * 2 + wblk];
            const v8bf* s = (const v8bf*)&sCent[lbl * BD];
            lo[rep] = s[0]; hi[rep] = s[1];
        }
    };
    auto storeX = [&](int buf, const float4* gx) {
#pragma unroll
        for (int i = 0; i < 4; ++i) {
            v4bf w;
            w[0] = (__bf16)gx[i].x; w[1] = (__bf16)gx[i].y;
            w[2] = (__bf16)gx[i].z; w[3] = (__bf16)gx[i].w;
            *(v4bf*)&sX[buf][xrow[i] * PITCH + xcol[i]] = w;
        }
    };
    auto storeW = [&](int buf, const v8bf* lo, const v8bf* hi) {
#pragma unroll
        for (int rep = 0; rep < 2; ++rep) {
            int n = rep * 128 + wrow;
            v8bf* d = (v8bf*)&sW[buf][n * PITCH + wblk * BD];
            d[0] = lo[rep]; d[1] = hi[rep];
        }
    };
    // bf16 16x32 fragment, documented per-lane layout:
    //   row = lane&15, K-base = (lane>>4)*8, chunks at K+0..7 and K+16..23
    auto ldfrag = [&](const __bf16* tbase) -> v16bf {
        const __bf16* p = tbase + (lane & 15) * PITCH + ((lane >> 4) << 3);
        v8bf lo = *(const v8bf*)p;
        v8bf hi = *(const v8bf*)(p + 16);
        return __builtin_shufflevector(lo, hi,
            0, 1, 2, 3, 4, 5, 6, 7, 8, 9, 10, 11, 12, 13, 14, 15);
    };

    v8f acc[4][4] = {};   // 4 M-tiles x 4 N-tiles of 16x16 f32 per wave

    // prologue: fill buffer 0
    {
        float4 gx[4]; v8bf lo[2], hi[2];
        fetchX(0, gx); fetchW(0, lo, hi);
        storeX(0, gx); storeW(0, lo, hi);
    }
    __syncthreads();

    for (int ks = 0; ks < NKSTEP; ++ks) {
        const int  buf  = ks & 1;
        const bool more = (ks + 1) < NKSTEP;
        float4 gx[4]; v8bf wlo[2], whi[2];

        if (more) {
            fetchX(ks + 1, gx);            // global loads in flight during WMMAs
            fetchW(ks + 1, wlo, whi);
            if (ks + 2 < NKSTEP) {         // warm k+2 lines (global_prefetch_b8)
                __builtin_prefetch(X + (size_t)(mbase + xrow[0]) * IN_F
                                     + (ks + 2) * KT + xcol[0], 0, 1);
                __builtin_prefetch(&L[(size_t)(nbase + wrow) * (IN_F / BD)
                                     + (ks + 2) * 2], 0, 1);
            }
        }

        // ---- matrix math on current buffer: 8 frags -> 16 WMMAs ----
        v16bf a[4], b[4];
#pragma unroll
        for (int i = 0; i < 4; ++i)
            a[i] = ldfrag(&sX[buf][(wm * 64 + i * 16) * PITCH]);
#pragma unroll
        for (int j = 0; j < 4; ++j)
            b[j] = ldfrag(&sW[buf][(wn * 64 + j * 16) * PITCH]);
#pragma unroll
        for (int i = 0; i < 4; ++i)
#pragma unroll
            for (int j = 0; j < 4; ++j)
                acc[i][j] = __builtin_amdgcn_wmma_f32_16x16x32_bf16(
                    false, a[i], false, b[j], (short)0, acc[i][j], false, false);

        if (more) { storeX(buf ^ 1, gx); storeW(buf ^ 1, wlo, whi); }
        __syncthreads();
    }

    // ---- epilogue: bias add + f32 store (half-wave -> 16 contiguous n) ----
    const int nlo   = lane & 15;
    const int mhalf = (lane >> 4) << 3;
#pragma unroll
    for (int j = 0; j < 4; ++j) {
        const int n  = nbase + wn * 64 + j * 16 + nlo;
        const float bv = B[n];
#pragma unroll
        for (int i = 0; i < 4; ++i) {
            const int m0 = mbase + wm * 64 + i * 16 + mhalf;
#pragma unroll
            for (int r = 0; r < 8; ++r)
                O[(size_t)(m0 + r) * OUT_F + n] = acc[i][j][r] + bv;
        }
    }
}

extern "C" void kernel_launch(void* const* d_in, const int* in_sizes, int n_in,
                              void* d_out, int out_size, void* d_ws, size_t ws_size,
                              hipStream_t stream) {
    const float* x    = (const float*)d_in[0];
    const float* cent = (const float*)d_in[1];
    const int*   lbl  = (const int*)d_in[2];
    const float* bias = (const float*)d_in[3];
    float*       out  = (float*)d_out;

    dim3 grid(OUT_F / NT, MROWS / MT);   // (16, 64)
    codebook_linear_wmma<<<grid, dim3(256), 0, stream>>>(x, cent, lbl, bias, out);

    (void)in_sizes; (void)n_in; (void)out_size; (void)d_ws; (void)ws_size;
}
